// SelfAttention_4406636446120
// MI455X (gfx1250) — compile-verified, matrix operand order checked by reading, not run on hardware
//
#include <hip/hip_runtime.h>

typedef __attribute__((ext_vector_type(16))) _Float16 v16h;
typedef __attribute__((ext_vector_type(8)))  float    v8f;
typedef __attribute__((ext_vector_type(4)))  float    v4f;
typedef __attribute__((vector_size(16)))     int      v4i_vs;

#define E_DIM 256
#define T_DIM 4096
#define B_DIM 4
#define N_HEADS 4
#define S_DIM 64
#define BT (B_DIM * T_DIM)

#if __has_builtin(__builtin_amdgcn_global_load_async_to_lds_b128) && \
    __has_builtin(__builtin_amdgcn_s_wait_asynccnt)
#define ASYNC_LDS 1
#endif

static __device__ __forceinline__ v8f wmma_f16(v16h a, v16h b, v8f c) {
  return __builtin_amdgcn_wmma_f32_16x16x32_f16(
      /*neg_a=*/false, a, /*neg_b=*/false, b,
      /*c_mod=*/(short)0, c, /*reuse_a=*/false, /*reuse_b=*/false);
}

// Load a 16x32 f16 A-fragment (or the layout-symmetric B-fragment) per the
// CDNA5 ISA VGPR tables: lane (l&15) owns row r0+(l&15); half-group selects
// K offsets {0..7,16..23} (lanes 0-15) or {8..15,24..31} (lanes 16-31).
// Two aligned 16-byte loads per lane. Works for global or LDS pointers.
static __device__ __forceinline__ v16h load_frag_g(const _Float16* __restrict__ base,
                                                   int ld, int r0, int k0) {
  int lane = threadIdx.x & 31;
  int r  = r0 + (lane & 15);
  int hi = (lane >> 4) & 1;
  const _Float16* p = base + (size_t)r * ld + k0 + hi * 8;
  union { v16h v; v4f q[2]; } u;
  u.q[0] = *(const v4f*)(p);
  u.q[1] = *(const v4f*)(p + 16);
  return u.v;
}

// 16-byte global -> LDS copy: async (ASYNCcnt-tracked, bypasses VGPRs) when
// the gfx1250 builtin is available, else staged through registers.
static __device__ __forceinline__ void async_cp16(const _Float16* g, _Float16* l) {
#if defined(ASYNC_LDS)
  __builtin_amdgcn_global_load_async_to_lds_b128(
      (__attribute__((address_space(1))) v4i_vs*)g,
      (__attribute__((address_space(3))) v4i_vs*)l, 0, 0);
#else
  *(v4f*)l = *(const v4f*)g;
#endif
}

static __device__ __forceinline__ void wait_async_lds() {
#if defined(ASYNC_LDS)
  __builtin_amdgcn_s_wait_asynccnt(0);
#endif
}

// ---------------------------------------------------------------- convert
__global__ void f32_to_f16_kernel(const float* __restrict__ src,
                                  _Float16* __restrict__ dst, int n) {
  int i = blockIdx.x * blockDim.x + threadIdx.x;
  if (i < n) dst[i] = (_Float16)src[i];
}

// ------------------------------------------------------- QKV projection GEMM
// Y[i][j] = sum_e X[i][e] * W[j][e]  (y = x @ W.T); W row-major already
// matches the B-fragment addressing (lane j reads W row j contiguously).
// mode 0/1: write [bh][t][s] layout (Q, K). mode 2: write [bh][s][t] (V^T).
__global__ void __launch_bounds__(128)
gemm_qkv_kernel(const _Float16* __restrict__ xh,
                const _Float16* __restrict__ wh,
                _Float16* __restrict__ dst, int mode) {
  int wave = threadIdx.x >> 5;
  int lane = threadIdx.x & 31;
  int hi   = (lane >> 4) & 1;
  int row0 = blockIdx.x * 16;
  int j0   = (blockIdx.y * 4 + wave) * 16;

  v8f acc = {};
#pragma unroll
  for (int kc = 0; kc < 8; ++kc) {
    v16h a = load_frag_g(xh, E_DIM, row0, kc * 32);
    v16h b = load_frag_g(wh, E_DIM, j0,   kc * 32);
    acc = wmma_f16(a, b, acc);
  }

  int j = j0 + (lane & 15);
  int h = j >> 6, s = j & 63;
#pragma unroll
  for (int vv = 0; vv < 8; ++vv) {
    int i  = row0 + vv + hi * 8;
    int bb = i >> 12, t = i & 4095;
    size_t idx;
    if (mode == 2)
      idx = ((size_t)((bb * N_HEADS + h) * S_DIM + s)) * T_DIM + t;
    else
      idx = ((size_t)((bb * N_HEADS + h) * T_DIM + t)) * S_DIM + s;
    dst[idx] = (_Float16)acc[vv];
  }
}

// ------------------------------------------------------------ flash attention
// 4 waves/block share one (b,h); each wave owns 16 query rows. K (32x64) and
// V^T (64x32) tiles are staged into LDS once per block via double-buffered
// async global->LDS copies; waves consume them as WMMA fragments from LDS.
// Online softmax across the 16-lane half-groups (C/D layout); P re-laid-out
// D->A through a per-wave LDS tile.
__global__ void __launch_bounds__(128)
flash_attn_kernel(const _Float16* __restrict__ qh,
                  const _Float16* __restrict__ kh,
                  const _Float16* __restrict__ vt,
                  _Float16* __restrict__ ao) {
  __shared__ _Float16 kbuf[2][32 * 64];   // [key j][s]
  __shared__ _Float16 vbuf[2][64 * 32];   // [s][key j]
  __shared__ _Float16 pbuf[4][16 * 32];

  int wave  = threadIdx.x >> 5;
  int lane  = threadIdx.x & 31;
  int hi    = (lane >> 4) & 1;
  int qtile = blockIdx.x * 4 + wave;
  int bh    = blockIdx.y;

  const _Float16* q = qh + (size_t)bh * T_DIM * S_DIM;
  const _Float16* k = kh + (size_t)bh * T_DIM * S_DIM;
  const _Float16* v = vt + (size_t)bh * S_DIM * T_DIM;
  int row0 = qtile * 16;

  // Q tile 16x64 held in registers for the whole sweep (2 A-fragments)
  v16h qa0 = load_frag_g(q, S_DIM, row0, 0);
  v16h qa1 = load_frag_g(q, S_DIM, row0, 32);

  v8f o0 = {}, o1 = {}, o2 = {}, o3 = {};
  float rowM[8], rowL[8];
#pragma unroll
  for (int i = 0; i < 8; ++i) { rowM[i] = -3.0e38f; rowL[i] = 0.0f; }
  const float scale = 0.125f;  // 1/sqrt(64)

  _Float16* pl = pbuf[wave];

  // cooperative tile loader: 128 threads x (2 K-chunks + 2 V-chunks) x 16B
  auto load_tiles = [&](int buf, int j0) {
    int tid = threadIdx.x;
#pragma unroll
    for (int c = 0; c < 2; ++c) {
      int idx = tid + c * 128;                  // 0..255
      int kr = idx >> 3, kc2 = (idx & 7) * 8;   // K: 32 rows x 8 chunks
      async_cp16(k + (size_t)(j0 + kr) * S_DIM + kc2, &kbuf[buf][kr * 64 + kc2]);
      int vr = idx >> 2, vc2 = (idx & 3) * 8;   // V^T: 64 rows x 4 chunks
      async_cp16(v + (size_t)vr * T_DIM + j0 + vc2, &vbuf[buf][vr * 32 + vc2]);
    }
  };

  load_tiles(0, 0);

  const int NIT = T_DIM / 32;  // 128
  for (int it = 0; it < NIT; ++it) {
    int cur = it & 1;
    wait_async_lds();
    __syncthreads();           // current tile resident; all waves done w/ other buffer
    if (it + 1 < NIT) load_tiles(cur ^ 1, (it + 1) * 32);

    const _Float16* kb = kbuf[cur];
    const _Float16* vb = vbuf[cur];

    // S = Q K^T for a 16x32 score tile (two 16x16 n-tiles, K=64 = 2x32)
    v8f s0 = {}, s1 = {};
    {
      v16h f = load_frag_g(kb, S_DIM, 0, 0);
      s0 = wmma_f16(qa0, f, s0);
      f = load_frag_g(kb, S_DIM, 0, 32);
      s0 = wmma_f16(qa1, f, s0);
      f = load_frag_g(kb, S_DIM, 16, 0);
      s1 = wmma_f16(qa0, f, s1);
      f = load_frag_g(kb, S_DIM, 16, 32);
      s1 = wmma_f16(qa1, f, s1);
    }

    // online softmax per row (row = slot + hi*8, spread over 16 lanes)
#pragma unroll
    for (int vv = 0; vv < 8; ++vv) {
      float a  = s0[vv] * scale;
      float bb = s1[vv] * scale;
      float m  = fmaxf(a, bb);
#pragma unroll
      for (int off = 1; off < 16; off <<= 1)
        m = fmaxf(m, __shfl_xor(m, off, 32));
      float mNew = fmaxf(rowM[vv], m);
      float corr = __expf(rowM[vv] - mNew);
      rowM[vv] = mNew;
      float p0 = __expf(a - mNew);
      float p1 = __expf(bb - mNew);
      float ls = p0 + p1;
#pragma unroll
      for (int off = 1; off < 16; off <<= 1)
        ls += __shfl_xor(ls, off, 32);
      rowL[vv] = rowL[vv] * corr + ls;
      o0[vv] *= corr; o1[vv] *= corr; o2[vv] *= corr; o3[vv] *= corr;
      int prow = vv + hi * 8;
      pl[prow * 32 + (lane & 15)]      = (_Float16)p0;
      pl[prow * 32 + 16 + (lane & 15)] = (_Float16)p1;
    }
    asm volatile("s_wait_dscnt 0x0" ::: "memory");

    // re-read P as a 16x32 A-fragment (D-layout -> A-layout via LDS)
    v16h pa;
    {
      int base = (lane & 15) * 32 + hi * 8;
#pragma unroll
      for (int t = 0; t < 8; ++t) pa[t] = pl[base + t];
#pragma unroll
      for (int t = 0; t < 8; ++t) pa[8 + t] = pl[base + 16 + t];
    }

    // O(16x64) += P(16x32) x V(32x64) from the shared V^T tile
    v16h f;
    f = load_frag_g(vb, 32, 0,  0); o0 = wmma_f16(pa, f, o0);
    f = load_frag_g(vb, 32, 16, 0); o1 = wmma_f16(pa, f, o1);
    f = load_frag_g(vb, 32, 32, 0); o2 = wmma_f16(pa, f, o2);
    f = load_frag_g(vb, 32, 48, 0); o3 = wmma_f16(pa, f, o3);
  }

  // epilogue: divide by rowsum, store f16 attention output [b*t][h*64+s]
  int b = bh >> 2, h = bh & 3;
#pragma unroll
  for (int vv = 0; vv < 8; ++vv) {
    int trow  = row0 + vv + hi * 8;
    float inv = 1.0f / rowL[vv];
    size_t base = ((size_t)(b * T_DIM + trow)) * E_DIM + h * S_DIM + (lane & 15);
    ao[base]      = (_Float16)(o0[vv] * inv);
    ao[base + 16] = (_Float16)(o1[vv] * inv);
    ao[base + 32] = (_Float16)(o2[vv] * inv);
    ao[base + 48] = (_Float16)(o3[vv] * inv);
  }
}

// ----------------------------------------------------- output projection GEMM
__global__ void __launch_bounds__(128)
gemm_out_kernel(const _Float16* __restrict__ ao,
                const _Float16* __restrict__ wuh,
                const float* __restrict__ bu,
                float* __restrict__ out) {
  int wave = threadIdx.x >> 5;
  int lane = threadIdx.x & 31;
  int hi   = (lane >> 4) & 1;
  int row0 = blockIdx.x * 16;
  int j0   = (blockIdx.y * 4 + wave) * 16;

  v8f acc = {};
#pragma unroll
  for (int kc = 0; kc < 8; ++kc) {
    v16h a = load_frag_g(ao,  E_DIM, row0, kc * 32);
    v16h b = load_frag_g(wuh, E_DIM, j0,   kc * 32);
    acc = wmma_f16(a, b, acc);
  }

  int j = j0 + (lane & 15);
  float bias = bu[j];
#pragma unroll
  for (int vv = 0; vv < 8; ++vv) {
    int i = row0 + vv + hi * 8;
    out[(size_t)i * E_DIM + j] = acc[vv] + bias;
  }
}

// ---------------------------------------------------------------------- launch
extern "C" void kernel_launch(void* const* d_in, const int* in_sizes, int n_in,
                              void* d_out, int out_size, void* d_ws, size_t ws_size,
                              hipStream_t stream) {
  const float* x  = (const float*)d_in[0];
  const float* Wq = (const float*)d_in[1];
  const float* Wk = (const float*)d_in[2];
  const float* Wv = (const float*)d_in[3];
  const float* Wu = (const float*)d_in[4];
  const float* bu = (const float*)d_in[5];

  char* ws = (char*)d_ws;
  size_t off = 0;
  _Float16* xh  = (_Float16*)(ws + off); off += (size_t)BT * E_DIM * 2;          // 8 MB
  _Float16* wqh = (_Float16*)(ws + off); off += (size_t)E_DIM * E_DIM * 2;
  _Float16* wkh = (_Float16*)(ws + off); off += (size_t)E_DIM * E_DIM * 2;
  _Float16* wvh = (_Float16*)(ws + off); off += (size_t)E_DIM * E_DIM * 2;
  _Float16* wuh = (_Float16*)(ws + off); off += (size_t)E_DIM * E_DIM * 2;
  _Float16* qhb = (_Float16*)(ws + off); off += (size_t)BT * E_DIM * 2;          // [bh][t][s]
  _Float16* khb = (_Float16*)(ws + off); off += (size_t)BT * E_DIM * 2;          // [bh][t][s]
  _Float16* vtb = (_Float16*)(ws + off); off += (size_t)BT * E_DIM * 2;          // [bh][s][t]
  _Float16* aob = (_Float16*)(ws + off); off += (size_t)BT * E_DIM * 2;          // [b*t][e]

  int nx = BT * E_DIM;
  int nw = E_DIM * E_DIM;
  f32_to_f16_kernel<<<(nx + 255) / 256, 256, 0, stream>>>(x,  xh,  nx);
  f32_to_f16_kernel<<<(nw + 255) / 256, 256, 0, stream>>>(Wq, wqh, nw);
  f32_to_f16_kernel<<<(nw + 255) / 256, 256, 0, stream>>>(Wk, wkh, nw);
  f32_to_f16_kernel<<<(nw + 255) / 256, 256, 0, stream>>>(Wv, wvh, nw);
  f32_to_f16_kernel<<<(nw + 255) / 256, 256, 0, stream>>>(Wu, wuh, nw);

  dim3 blk(128);
  dim3 gGemm(BT / 16, 4);  // 1024 row tiles x (4 waves x 4) col tiles
  gemm_qkv_kernel<<<gGemm, blk, 0, stream>>>(xh, wqh, qhb, 0);
  gemm_qkv_kernel<<<gGemm, blk, 0, stream>>>(xh, wkh, khb, 1);
  gemm_qkv_kernel<<<gGemm, blk, 0, stream>>>(xh, wvh, vtb, 2);

  dim3 gAttn(T_DIM / 16 / 4, B_DIM * N_HEADS);  // (64, 16), 4 waves/block
  flash_attn_kernel<<<gAttn, blk, 0, stream>>>(qhb, khb, vtb, aob);

  gemm_out_kernel<<<gGemm, blk, 0, stream>>>(aob, wuh, bu, (float*)d_out);
}
